// EnergyAwareTTTRouter_5059471475439
// MI455X (gfx1250) — compile-verified, matrix operand order checked by reading, not run on hardware
//
#include <hip/hip_runtime.h>

typedef __attribute__((ext_vector_type(16))) __bf16 v16bf;
typedef __attribute__((ext_vector_type(8)))  float  v8f;

#define D_MODEL     2048
#define NUM_EXPERTS 64
#define TOKENS      16384
#define KSTEPS      (D_MODEL / 32)     // 64 WMMA K-steps
#define N_TILES     4                  // 64 experts / 16
#define M_TILES     2                  // 32 tokens per wave
#define WPB         4                  // waves per block
#define WS_B_OFF    1024               // byte offset of packed B in d_ws

// ---------------- kernel 1: penalties + zero histogram ----------------
__global__ void router_prep(const float* __restrict__ usage,
                            const float* __restrict__ bias,
                            float* __restrict__ pen,
                            float* __restrict__ usage_out)
{
    int e = threadIdx.x;
    if (e >= NUM_EXPERTS) return;
    float s = 0.f, mx = 0.f;
    for (int i = 0; i < NUM_EXPERTS; ++i) {
        float u = usage[i];
        s += u;
        mx = fmaxf(mx, u);
    }
    float inv  = 1.f / (s + 1e-8f);
    float rmax = mx * inv;
    float norm = (rmax > 0.f) ? (usage[e] * inv) / rmax : 0.f;
    const float base = 0.001f * 1000.f * 0.5f;  // LAMBDA * SCALE * LAST_ENERGY
    pen[e]       = base * (0.1f + 1.9f * norm) - bias[e];  // folds (zero) bias
    usage_out[e] = 0.f;
}

// ------- kernel 2: pre-pack W -> bf16 WMMA B-fragments (ISA 32x16 layout) -------
// record (kstep, t, lane): lane holds column n = lane&15, group = lane>>4,
// element idx in 0..15 maps to K = kstep*32 + idx + grp*8 + (idx>=8 ? 8 : 0)
__global__ void router_pack_w(const float* __restrict__ W, v16bf* __restrict__ Bp)
{
    int gid = blockIdx.x * blockDim.x + threadIdx.x;
    if (gid >= KSTEPS * N_TILES * 32) return;
    int lane  = gid & 31;
    int rec   = gid >> 5;
    int t     = rec & 3;
    int kstep = rec >> 2;
    int grp   = lane >> 4;
    int n     = lane & 15;
    v16bf b;
#pragma unroll
    for (int idx = 0; idx < 16; ++idx) {
        int K = kstep * 32 + idx + grp * 8 + (idx >= 8 ? 8 : 0);
        b[idx] = (__bf16)W[K * NUM_EXPERTS + t * 16 + n];
    }
    Bp[gid] = b;
}

// ---------------- kernel 3: WMMA GEMM + fused softmax/top-8 ----------------
// One wave owns 32 tokens (2 M-tiles): each B fragment feeds 2 WMMAs, and in
// the epilogue every lane owns exactly one token (full wave32 utilization).
__global__ __launch_bounds__(WPB * 32)
void router_main(const float* __restrict__ x,
                 const float* __restrict__ pen,
                 const v16bf* __restrict__ Bp,
                 float* __restrict__ out)
{
    __shared__ float lds[WPB][32 * NUM_EXPERTS];   // 32 KB / block

    const int lane = threadIdx.x & 31;
    const int w    = threadIdx.x >> 5;
    const int blk  = blockIdx.x * WPB + w;         // 512 row-blocks of 32 tokens
    const int row0 = blk * 32;
    const int grp  = lane >> 4;
    const int m    = lane & 15;

    const float* xrow0 = x + (size_t)(row0 + m) * D_MODEL;        // M-tile 0
    const float* xrow1 = x + (size_t)(row0 + 16 + m) * D_MODEL;   // M-tile 1

    v8f acc[M_TILES][N_TILES] = {};

#pragma unroll 1
    for (int ks = 0; ks < KSTEPS; ++ks) {
        const int k0 = ks * 32;
        if (k0 + 512 < D_MODEL) {
            __builtin_prefetch(xrow0 + k0 + 512, 0, 1);   // global_prefetch_b8
            __builtin_prefetch(xrow1 + k0 + 512, 0, 1);
        }

        // A fragments: ISA 16-bit A 16x32 layout (pairs contiguous -> b128-merged)
        v16bf a[M_TILES];
#pragma unroll
        for (int h = 0; h < M_TILES; ++h) {
            const float* xr = (h == 0) ? xrow0 : xrow1;
#pragma unroll
            for (int v = 0; v < 8; ++v) {
                int kb   = 2 * v + grp * 8 + (2 * v >= 8 ? 8 : 0);
                float2 f = *(const float2*)(xr + k0 + kb);
                a[h][2 * v]     = (__bf16)f.x;
                a[h][2 * v + 1] = (__bf16)f.y;
            }
        }

        v16bf b[N_TILES];
#pragma unroll
        for (int t = 0; t < N_TILES; ++t)
            b[t] = Bp[(ks * N_TILES + t) * 32 + lane];

#pragma unroll
        for (int t = 0; t < N_TILES; ++t) {
            acc[0][t] = __builtin_amdgcn_wmma_f32_16x16x32_bf16(
                false, a[0], false, b[t], (short)0, acc[0][t], false, false);
            acc[1][t] = __builtin_amdgcn_wmma_f32_16x16x32_bf16(
                false, a[1], false, b[t], (short)0, acc[1][t], false, false);
        }
    }

    // C/D layout: VGPR j -> row (j + grp*8), lane%16 -> column within N-tile
    float pe[N_TILES];
#pragma unroll
    for (int t = 0; t < N_TILES; ++t) pe[t] = pen[t * 16 + m];

#pragma unroll
    for (int h = 0; h < M_TILES; ++h)
#pragma unroll
        for (int t = 0; t < N_TILES; ++t)
#pragma unroll
            for (int j = 0; j < 8; ++j)
                lds[w][(h * 16 + j + grp * 8) * NUM_EXPERTS + t * 16 + m] =
                    acc[h][t][j] - pe[t];

    __syncthreads();

    // -------- every lane owns one token --------
    const int token = row0 + lane;

    float v[NUM_EXPERTS];
#pragma unroll
    for (int e = 0; e < NUM_EXPERTS; ++e) v[e] = lds[w][lane * NUM_EXPERTS + e];

    float mx = v[0];
#pragma unroll
    for (int e = 1; e < NUM_EXPERTS; ++e) mx = fmaxf(mx, v[e]);

    float s = 0.f;
    unsigned key[NUM_EXPERTS];
#pragma unroll
    for (int e = 0; e < NUM_EXPERTS; ++e) {
        float p = __expf(v[e] - mx);
        s += p;
        // p > 0 => float bits are order-preserving as u32.
        // Mask low 6 mantissa bits, embed complemented index: on ties the
        // max picks the LOWEST expert index (matches jax.lax.top_k).
        key[e] = (__float_as_uint(p) & 0xFFFFFFC0u) | (unsigned)(63 - e);
    }

    float ts = 0.f;
    unsigned bestk[8];
#pragma unroll
    for (int kk = 0; kk < 8; ++kk) {
        unsigned best = 0u;
#pragma unroll
        for (int e = 0; e < NUM_EXPERTS; ++e) best = max(best, key[e]);
        bestk[kk] = best;
        ts += __uint_as_float(best & 0xFFFFFFC0u);
#pragma unroll
        for (int e = 0; e < NUM_EXPERTS; ++e)
            if (key[e] == best) key[e] = 0u;      // keys unique: removes exactly one
    }

    // probs_topk = (p_i/s) / (sum_top/s + 1e-8)  ==  p_i / (ts + 1e-8*s)
    float inv = 1.f / (ts + 1e-8f * s);
#pragma unroll
    for (int kk = 0; kk < 8; ++kk) {
        int   bi = 63 - (int)(bestk[kk] & 63u);
        float bv = __uint_as_float(bestk[kk] & 0xFFFFFFC0u);
        out[token * 8 + kk]              = (float)bi;        // indices (flattened as float)
        out[TOKENS * 8 + token * 8 + kk] = bv * inv;         // renormalized probs
        atomicAdd(&out[2 * TOKENS * 8 + bi], 1.f);           // expert_usage histogram
    }
}

extern "C" void kernel_launch(void* const* d_in, const int* in_sizes, int n_in,
                              void* d_out, int out_size, void* d_ws, size_t ws_size,
                              hipStream_t stream)
{
    const float* x     = (const float*)d_in[0];
    const float* W     = (const float*)d_in[1];
    const float* b     = (const float*)d_in[2];
    const float* usage = (const float*)d_in[3];
    float* out = (float*)d_out;

    float* pen = (float*)d_ws;
    v16bf* Bp  = (v16bf*)((char*)d_ws + WS_B_OFF);   // 256 KB packed B (L2-resident)

    router_prep<<<1, 64, 0, stream>>>(usage, b, pen, out + 2 * TOKENS * 8);

    router_pack_w<<<(KSTEPS * N_TILES * 32 + 255) / 256, 256, 0, stream>>>(W, Bp);

    router_main<<<(TOKENS / 32) / WPB, WPB * 32, 0, stream>>>(x, pen, Bp, out);
}
